// GCN_36412732735978
// MI455X (gfx1250) — matrix operand order repaired and checked by reading, not compiled, not run on hardware
//
#include <hip/hip_runtime.h>
#include <hip/hip_bf16.h>

#define N_NODES   50000
#define N_EDGES   800000
#define IN_DIM    128
#define HGC       128
#define HID2      256
#define NCLS      10
#define NFEAT     (N_NODES * IN_DIM)

typedef __attribute__((ext_vector_type(16))) __bf16 v16bf;
typedef __attribute__((ext_vector_type(8)))  float  v8f;

// ---------------- utility kernels ----------------

__global__ void zero_f32_kernel(float* __restrict__ p, int n) {
    int i = blockIdx.x * blockDim.x + threadIdx.x;
    if (i < n) p[i] = 0.0f;
}

// deg[row[e]] += ew[e]
__global__ void deg_kernel(const int* __restrict__ row, const float* __restrict__ ew,
                           float* __restrict__ deg, int E) {
    int e = blockIdx.x * blockDim.x + threadIdx.x;
    if (e < E) atomicAdd(&deg[row[e]], ew[e]);
}

// w[e] = -dinv[row]*ew*dinv[col]; also passthrough edge_weight into output tail
__global__ void lapw_kernel(const int* __restrict__ row, const int* __restrict__ col,
                            const float* __restrict__ ew, const float* __restrict__ deg,
                            float* __restrict__ w, float* __restrict__ ewOut, int E) {
    int e = blockIdx.x * blockDim.x + threadIdx.x;
    if (e >= E) return;
    float dr = deg[row[e]];
    float dc = deg[col[e]];
    float ir = (dr > 0.0f) ? rsqrtf(fmaxf(dr, 1e-30f)) : 0.0f;
    float ic = (dc > 0.0f) ? rsqrtf(fmaxf(dc, 1e-30f)) : 0.0f;
    float v  = ew[e];
    w[e] = -ir * v * ic;
    ewOut[e] = v;
}

// edge-parallel SpMM: y[row[e],:] += w[e] * x[col[e],:]   (one wave32 per edge, 4 feats/lane)
__global__ void spmm_atomic_kernel(const int* __restrict__ row, const int* __restrict__ col,
                                   const float* __restrict__ w, const float* __restrict__ x,
                                   float* __restrict__ y, int E) {
    int t = blockIdx.x * blockDim.x + threadIdx.x;
    int e = t >> 5;
    int lane = t & 31;
    if (e >= E) return;
    float wv = w[e];
    const float* xs = x + (size_t)col[e] * IN_DIM;
    float*       yd = y + (size_t)row[e] * IN_DIM;
#pragma unroll
    for (int j = 0; j < 4; ++j) {
        int f = lane + 32 * j;
        atomicAdd(&yd[f], wv * xs[f]);
    }
}

// z = 2*z - t0  (Chebyshev T2)
__global__ void cheb2_kernel(float* __restrict__ z, const float* __restrict__ t0, int n) {
    int i = blockIdx.x * blockDim.x + threadIdx.x;
    if (i < n) z[i] = 2.0f * z[i] - t0[i];
}

// Pack fp32 weight (Ksrc x srcStride row-major, Ncols valid) into bf16 WMMA-B layout.
// dst index = ((kcIdx*(Npad/16) + nt)*32 + lane)*16 + j ; element = W[kcIdx*32 + (lane/16)*16 + j][nt*16 + lane%16]
__global__ void pack_b_bf16_kernel(const float* __restrict__ src, __bf16* __restrict__ dst,
                                   int Ksrc, int srcStride, int Ncols, int Npad) {
    int total = (Ksrc / 32) * (Npad / 16) * 512;
    int idx = blockIdx.x * blockDim.x + threadIdx.x;
    if (idx >= total) return;
    int j    = idx & 15;
    int lane = (idx >> 4) & 31;
    int tile = idx >> 9;
    int ntiles = Npad / 16;
    int kcIdx = tile / ntiles;
    int nt    = tile % ntiles;
    int k = kcIdx * 32 + (lane >> 4) * 16 + j;
    int n = nt * 16 + (lane & 15);
    float v = (n < Ncols) ? src[(size_t)k * srcStride + n] : 0.0f;
    dst[idx] = (__bf16)v;
}

// ---------------- WMMA GEMM with async double-buffered LDS staging of B ----------------
// C[M x nCols] = relu?( [A0|A1|A2](M x nParts*128) @ packedB + bias )
// 256 threads = 8 waves; each wave does 16 rows x (NT*16) cols. Block tile: 128 x NT*16.
// B K-chunks (NT KB each) are staged into LDS with GLOBAL_LOAD_ASYNC_TO_LDS_B128
// (ASYNCcnt-tracked) one chunk ahead, so WMMAs on chunk k overlap the fetch of chunk k+1.
template <int NT>
__global__ void wmma_gemm_kernel(const float* __restrict__ a0, const float* __restrict__ a1,
                                 const float* __restrict__ a2, int lda, int nParts,
                                 const __bf16* __restrict__ packedB,
                                 const float* __restrict__ bias, float* __restrict__ C,
                                 int M, int ldc, int nCols, int relu) {
    __shared__ __bf16 Bs[2][NT * 512];
    const int CHUNK_BYTES = NT * 1024;  // one K-chunk of packed B (NT tiles * 32 lanes * 16 bf16)

    int wave = threadIdx.x >> 5;
    int lane = threadIdx.x & 31;
    int mBase = blockIdx.x * 128 + wave * 16;
    int r = mBase + (lane & 15);
    int khalf = lane >> 4;

    uint64_t pb = (uint64_t)packedB;                      // uniform -> SGPR pair
    unsigned ldsBase[2];
    ldsBase[0] = (unsigned)(size_t)(&Bs[0][0]);           // low 32 bits = LDS byte offset
    ldsBase[1] = (unsigned)(size_t)(&Bs[1][0]);

    // prologue: async-stage chunk 0 into buffer 0
    for (int o = (int)threadIdx.x * 16; o < CHUNK_BYTES; o += 256 * 16) {
        unsigned lds  = ldsBase[0] + (unsigned)o;
        unsigned goff = (unsigned)o;
        asm volatile("global_load_async_to_lds_b128 %0, %1, %2"
                     :: "v"(lds), "v"(goff), "s"(pb) : "memory");
    }
    asm volatile("s_wait_asynccnt 0x0" ::: "memory");
    __syncthreads();

    v8f acc[NT];
#pragma unroll
    for (int t = 0; t < NT; ++t)
#pragma unroll
        for (int i = 0; i < 8; ++i) acc[t][i] = 0.0f;

    int nChunks = nParts * 4;  // K chunks of 32
    for (int kc = 0; kc < nChunks; ++kc) {
        int cur = kc & 1;

        // async-prefetch next chunk into the other buffer (consumed chunk kc-1; barrier passed)
        if (kc + 1 < nChunks) {
            unsigned gbase = (unsigned)(kc + 1) * (unsigned)CHUNK_BYTES;
            unsigned lbase = ldsBase[cur ^ 1];
            for (int o = (int)threadIdx.x * 16; o < CHUNK_BYTES; o += 256 * 16) {
                unsigned lds  = lbase + (unsigned)o;
                unsigned goff = gbase + (unsigned)o;
                asm volatile("global_load_async_to_lds_b128 %0, %1, %2"
                             :: "v"(lds), "v"(goff), "s"(pb) : "memory");
            }
        }

        // A fragment: fp32 global loads -> bf16 (contiguous k = 16*khalf + i, same order as packed B)
        int part = kc >> 2;
        const float* ap = (part == 0) ? a0 : (part == 1) ? a1 : a2;
        int kk = (kc & 3) * 32;

        float fv[16];
        if (r < M) {
            const float* p = ap + (size_t)r * lda + kk + khalf * 16;
            float4 x0 = *(const float4*)(p);
            float4 x1 = *(const float4*)(p + 4);
            float4 y0 = *(const float4*)(p + 8);
            float4 y1 = *(const float4*)(p + 12);
            fv[0] = x0.x; fv[1] = x0.y; fv[2]  = x0.z; fv[3]  = x0.w;
            fv[4] = x1.x; fv[5] = x1.y; fv[6]  = x1.z; fv[7]  = x1.w;
            fv[8] = y0.x; fv[9] = y0.y; fv[10] = y0.z; fv[11] = y0.w;
            fv[12] = y1.x; fv[13] = y1.y; fv[14] = y1.z; fv[15] = y1.w;
        } else {
#pragma unroll
            for (int i = 0; i < 16; ++i) fv[i] = 0.0f;
        }
        v16bf afrag;
#pragma unroll
        for (int i = 0; i < 16; ++i) afrag[i] = (__bf16)fv[i];

        // B fragments from LDS (ds_load_b128 pairs), one WMMA per 16-col tile
#pragma unroll
        for (int nt = 0; nt < NT; ++nt) {
            v16bf bfrag = *(const v16bf*)(&Bs[cur][(size_t)nt * 512 + (size_t)lane * 16]);
            acc[nt] = __builtin_amdgcn_wmma_f32_16x16x32_bf16(
                false, afrag, false, bfrag, (short)0, acc[nt], false, false);
        }

        // next chunk fully resident + all waves done with 'cur' before it is overwritten
        asm volatile("s_wait_asynccnt 0x0" ::: "memory");
        __syncthreads();
    }

    int colBase = lane & 15;
    int rowOff = khalf * 8;
#pragma unroll
    for (int nt = 0; nt < NT; ++nt) {
        int col = nt * 16 + colBase;
        float b = (bias != nullptr && col < nCols) ? bias[col] : 0.0f;
#pragma unroll
        for (int v = 0; v < 8; ++v) {
            int rw = mBase + rowOff + v;
            if (rw < M && col < nCols) {
                float val = acc[nt][v] + b;
                if (relu) val = fmaxf(val, 0.0f);
                C[(size_t)rw * ldc + col] = val;
            }
        }
    }
}

// ---------------- BatchNorm stats + fold ----------------

__global__ void bn_stats_kernel(const float* __restrict__ h, const float* __restrict__ gamma,
                                float* __restrict__ mean, float* __restrict__ sfac, int M) {
    __shared__ float s1[256];
    __shared__ float s2[256];
    int ch = blockIdx.x;
    float a = 0.0f, b = 0.0f;
    for (int r = threadIdx.x; r < M; r += 256) {
        float v = h[(size_t)r * HID2 + ch];
        a += v; b += v * v;
    }
    s1[threadIdx.x] = a; s2[threadIdx.x] = b;
    __syncthreads();
    for (int st = 128; st > 0; st >>= 1) {
        if ((int)threadIdx.x < st) {
            s1[threadIdx.x] += s1[threadIdx.x + st];
            s2[threadIdx.x] += s2[threadIdx.x + st];
        }
        __syncthreads();
    }
    if (threadIdx.x == 0) {
        float m = s1[0] / (float)M;
        float var = s2[0] / (float)M - m * m;
        mean[ch] = m;
        sfac[ch] = gamma[ch] * rsqrtf(var + 1e-5f);
    }
}

__global__ void bn_foldw_kernel(const float* __restrict__ Wc2, const float* __restrict__ sfac,
                                float* __restrict__ w2fold) {
    int idx = blockIdx.x * blockDim.x + threadIdx.x;
    if (idx >= HID2 * 16) return;
    int k = idx >> 4, n = idx & 15;
    w2fold[idx] = (n < NCLS) ? sfac[k] * Wc2[k * NCLS + n] : 0.0f;
}

__global__ void bn_foldb_kernel(const float* __restrict__ Wc2, const float* __restrict__ bc2,
                                const float* __restrict__ beta, const float* __restrict__ mean,
                                const float* __restrict__ sfac, float* __restrict__ b2fold) {
    int n = threadIdx.x;
    if (n >= 16) return;
    float b = (n < NCLS) ? bc2[n] : 0.0f;
    if (n < NCLS) {
        for (int k = 0; k < HID2; ++k)
            b += (beta[k] - mean[k] * sfac[k]) * Wc2[k * NCLS + n];
    }
    b2fold[n] = b;
}

// ---------------- launcher ----------------

extern "C" void kernel_launch(void* const* d_in, const int* in_sizes, int n_in,
                              void* d_out, int out_size, void* d_ws, size_t ws_size,
                              hipStream_t stream) {
    (void)in_sizes; (void)n_in; (void)out_size; (void)ws_size;

    const float* feat  = (const float*)d_in[0];
    const int*   ei    = (const int*)d_in[1];      // [2, E]
    const float* ew    = (const float*)d_in[2];
    const float* W0    = (const float*)d_in[3];    // (3,128,128) contiguous -> (384,128)
    const float* W1    = (const float*)d_in[4];
    const float* W2    = (const float*)d_in[5];
    const float* Wc1   = (const float*)d_in[6];    // (128,256)
    const float* bc1   = (const float*)d_in[7];
    const float* gamma = (const float*)d_in[8];
    const float* beta  = (const float*)d_in[9];
    const float* Wc2   = (const float*)d_in[10];   // (256,10)
    const float* bc2   = (const float*)d_in[11];

    const int* row = ei;
    const int* col = ei + N_EDGES;

    float* out = (float*)d_out;               // logits [50000*10] then edge_weight [800000]
    float* ewOut = out + (size_t)N_NODES * NCLS;

    // workspace carve-out (256B aligned slabs)
    uint8_t* ws = (uint8_t*)d_ws;
    size_t off = 0;
    auto carve = [&](size_t bytes) -> uint8_t* {
        uint8_t* p = ws + off;
        off += (bytes + 255) & ~(size_t)255;
        return p;
    };
    float*  deg     = (float*)carve((size_t)N_NODES * 4);
    float*  wbuf    = (float*)carve((size_t)N_EDGES * 4);
    float*  X0      = (float*)carve((size_t)NFEAT * 4);
    float*  B1      = (float*)carve((size_t)NFEAT * 4);   // B1,B2 contiguous: h aliases them
    float*  B2      = (float*)carve((size_t)NFEAT * 4);
    float*  B3      = (float*)carve((size_t)NFEAT * 4);
    __bf16* packedW = (__bf16*)carve(131072);
    float*  meanb   = (float*)carve(HID2 * 4);
    float*  sfac    = (float*)carve(HID2 * 4);
    float*  w2fold  = (float*)carve(HID2 * 16 * 4);
    float*  b2fold  = (float*)carve(16 * 4);
    float*  h       = B1;                                  // 50000 x 256 spans B1+B2

    const int TB = 256;
    const int gE     = (N_EDGES + TB - 1) / TB;
    const int gE32   = (N_EDGES * 32 + TB - 1) / TB;
    const int gFeat  = (NFEAT + TB - 1) / TB;
    const int gNodes = (N_NODES + TB - 1) / TB;
    const int gGemm  = (N_NODES + 127) / 128;

    // 1) scaled Laplacian weights
    zero_f32_kernel<<<gNodes, TB, 0, stream>>>(deg, N_NODES);
    deg_kernel<<<gE, TB, 0, stream>>>(row, ew, deg, N_EDGES);
    lapw_kernel<<<gE, TB, 0, stream>>>(row, col, ew, deg, wbuf, ewOut, N_EDGES);

    // 2) three ChebConv layers: out = relu(T0@Wk0 + T1@Wk1 + T2@Wk2)
    auto run_layer = [&](const float* in, float* lout, const float* Wstack) {
        zero_f32_kernel<<<gFeat, TB, 0, stream>>>(B1, NFEAT);
        spmm_atomic_kernel<<<gE32, TB, 0, stream>>>(row, col, wbuf, in, B1, N_EDGES);   // T1 = L x
        zero_f32_kernel<<<gFeat, TB, 0, stream>>>(B2, NFEAT);
        spmm_atomic_kernel<<<gE32, TB, 0, stream>>>(row, col, wbuf, B1, B2, N_EDGES);   // L T1
        cheb2_kernel<<<gFeat, TB, 0, stream>>>(B2, in, NFEAT);                          // T2 = 2 L T1 - T0
        pack_b_bf16_kernel<<<(12 * 8 * 512 + TB - 1) / TB, TB, 0, stream>>>(
            Wstack, packedW, 384, HGC, HGC, HGC);
        wmma_gemm_kernel<8><<<gGemm, TB, 0, stream>>>(
            in, B1, B2, IN_DIM, 3, packedW, nullptr, lout, N_NODES, HGC, HGC, 1);
    };
    run_layer(feat, X0, W0);
    run_layer(X0, B3, W1);
    run_layer(B3, X0, W2);

    // 3) classifier layer 1: h = relu(x @ Wc1 + bc1)   (h overwrites B1/B2)
    pack_b_bf16_kernel<<<(4 * 16 * 512 + TB - 1) / TB, TB, 0, stream>>>(
        Wc1, packedW, HGC, HID2, HID2, HID2);
    wmma_gemm_kernel<16><<<gGemm, TB, 0, stream>>>(
        X0, X0, X0, HGC, 1, packedW, bc1, h, N_NODES, HID2, HID2, 1);

    // 4) BatchNorm folded into final linear: logit = h @ (s*Wc2) + b'
    bn_stats_kernel<<<HID2, TB, 0, stream>>>(h, gamma, meanb, sfac, N_NODES);
    bn_foldw_kernel<<<(HID2 * 16 + TB - 1) / TB, TB, 0, stream>>>(Wc2, sfac, w2fold);
    bn_foldb_kernel<<<1, 32, 0, stream>>>(Wc2, bc2, beta, meanb, sfac, b2fold);
    pack_b_bf16_kernel<<<(8 * 1 * 512 + TB - 1) / TB, TB, 0, stream>>>(
        w2fold, packedW, HID2, 16, 16, 16);

    // 5) final GEMM 50000x256 @ 256x16 -> logits (store 10 cols)
    wmma_gemm_kernel<1><<<gGemm, TB, 0, stream>>>(
        h, h + 128, h, HID2, 2, packedW, b2fold, out, N_NODES, NCLS, NCLS, 0);
}
